// GroupedQueryAttention_86629490360600
// MI455X (gfx1250) — matrix operand order repaired and checked by reading, not compiled, not run on hardware
//
#include <hip/hip_runtime.h>
#include <hip/hip_bf16.h>
#include <math.h>
#include <stdint.h>

typedef __attribute__((ext_vector_type(16))) __bf16 v16bf;
typedef __attribute__((ext_vector_type(8)))  __bf16 v8bf;
typedef __attribute__((ext_vector_type(8)))  float  v8f;
typedef __attribute__((ext_vector_type(4)))  int    v4i;

#define D_MODEL 1024
#define N_HEADS 16
#define D_HEAD  64
#define N_GROUPS 8
#define SEQ   2048
#define BATCH 2
#define BS    (BATCH*SEQ)   // 4096 tokens
#define NQKV  2048          // fused Q|K|V columns
#define KOFF  1024
#define VOFF  1536

// ---- CDNA5 async global->LDS copy (guarded; falls back to reg bounce) ------

#if defined(__HIP_DEVICE_COMPILE__) &&                                        \
    __has_builtin(__builtin_amdgcn_global_load_async_to_lds_b128) &&          \
    __has_builtin(__builtin_amdgcn_s_wait_asynccnt)
#define HAS_ASYNC_LDS 1
#else
#define HAS_ASYNC_LDS 0
#endif

#if HAS_ASYNC_LDS
typedef __attribute__((address_space(1))) v4i as1_v4i;
typedef __attribute__((address_space(3))) v4i as3_v4i;
#define ASYNC_CP16(g, l)                                                      \
  __builtin_amdgcn_global_load_async_to_lds_b128(                             \
      (as1_v4i*)(uintptr_t)(const void*)(g),                                  \
      (as3_v4i*)(uint32_t)(uintptr_t)(const void*)(l),                        \
      0, 0)
#define WAIT_ASYNC(n) __builtin_amdgcn_s_wait_asynccnt(n)
#else
#define ASYNC_CP16(g, l) (*(v8bf*)(l) = *(const v8bf*)(g))
#define WAIT_ASYNC(n) ((void)0)
#endif

// ---- WMMA helpers ---------------------------------------------------------

static __device__ __forceinline__ v8f wmma_bf16(v16bf a, v16bf b, v8f c) {
  // D = A(16x32 bf16) x B(32x16 bf16) + C(16x16 f32)
  return __builtin_amdgcn_wmma_f32_16x16x32_bf16(false, a, false, b, (short)0, c,
                                                 false, false);
}

// A-fragment (16x32 bf16, row-major source): lane holds row M=lane&15.
// Lanes 0-15 hold K={0..15}; lanes 16-31 hold K={16..31}.
static __device__ __forceinline__ v16bf load_a_frag(const __bf16* p, int kgrp) {
  return *(const v16bf*)(p + 2 * kgrp);
}

static __device__ __forceinline__ float rmax16(float v) {
  #pragma unroll
  for (int i = 1; i < 16; i <<= 1) v = fmaxf(v, __shfl_xor(v, i));
  return v;
}
static __device__ __forceinline__ float rsum16(float v) {
  #pragma unroll
  for (int i = 1; i < 16; i <<= 1) v += __shfl_xor(v, i);
  return v;
}

// ---- precision / layout prep ----------------------------------------------

__global__ void k_cvt_resid(const float* __restrict__ x, __bf16* __restrict__ y, int n) {
  int i = blockIdx.x * blockDim.x + threadIdx.x;
  if (i < n) y[i] = (__bf16)x[i];
}

__global__ void k_build_wcatT(const float* __restrict__ WQ, const float* __restrict__ WK,
                              const float* __restrict__ WV, __bf16* __restrict__ WcatT) {
  int idx = blockIdx.x * blockDim.x + threadIdx.x;       // n*1024 + d
  int d = idx & 1023;
  int n = idx >> 10;
  float v;
  if (n < 1024)      v = WQ[(size_t)d * 1024 + n];
  else if (n < 1536) v = WK[(size_t)d * 512 + (n - 1024)];
  else               v = WV[(size_t)d * 512 + (n - 1536)];
  WcatT[idx] = (__bf16)v;
}

__global__ void k_build_woT(const float* __restrict__ Wout, __bf16* __restrict__ WoT) {
  int idx = blockIdx.x * blockDim.x + threadIdx.x;       // d*1024 + (h*64+e)
  int c = idx & 1023;
  int d = idx >> 10;
  int h = c >> 6;
  int e = c & 63;
  WoT[idx] = (__bf16)Wout[((size_t)(e * 16 + h)) * 1024 + d];
}

__global__ void k_transpose_v(const __bf16* __restrict__ QKV, __bf16* __restrict__ Vt) {
  int idx = blockIdx.x * blockDim.x + threadIdx.x;       // ((b*8+g)*64+e)*2048 + t
  int t = idx & (SEQ - 1);
  int e = (idx >> 11) & 63;
  int g = (idx >> 17) & 7;
  int b = idx >> 20;
  Vt[idx] = QKV[((size_t)(b * SEQ + t)) * NQKV + VOFF + g * 64 + e];
}

// ---- fused QKV projection GEMM with async double-buffered B tiles ----------

__global__ __launch_bounds__(256) void k_gemm_qkv(const __bf16* __restrict__ A,
                                                  const __bf16* __restrict__ BT,
                                                  __bf16* __restrict__ C) {
  __shared__ alignas(32) __bf16 ldsB[2 * 64 * 32];        // 2 x 4 KB
  const int tid = threadIdx.x;
  const int lane = tid & 31, wid = tid >> 5;
  const int hi = lane >> 4, lm = lane & 15, kgrp = hi * 8;
  const int m0 = blockIdx.y * 128 + wid * 16;
  const int n0 = blockIdx.x * 64;
  // staging: 256 chunks of 16 B (64 n-rows x 4 chunks)
  const int sn = tid >> 2, sc = (tid & 3) * 8;
  const __bf16* bsrc = BT + (size_t)(n0 + sn) * D_MODEL + sc;

  v8f acc[4] = {};
  const __bf16* arow = A + (size_t)(m0 + lm) * D_MODEL;

  ASYNC_CP16(bsrc, ldsB + sn * 32 + sc);                  // k-step 0 -> buf 0
  for (int i = 0; i < D_MODEL / 32; ++i) {
    WAIT_ASYNC(0);
    __syncthreads();                                      // buf[i&1] ready; buf[(i+1)&1] free
    if (i + 1 < D_MODEL / 32)
      ASYNC_CP16(bsrc + (i + 1) * 32,
                 ldsB + ((i + 1) & 1) * 2048 + sn * 32 + sc);
    const __bf16* bufB = ldsB + (i & 1) * 2048;
    v16bf af = load_a_frag(arow + i * 32, kgrp);
    #pragma unroll
    for (int nt = 0; nt < 4; ++nt) {
      v16bf bf = *(const v16bf*)(bufB + (nt * 16 + lm) * 32 + hi * 16);
      acc[nt] = wmma_bf16(af, bf, acc[nt]);
    }
  }
  #pragma unroll
  for (int nt = 0; nt < 4; ++nt)
    #pragma unroll
    for (int r = 0; r < 8; ++r)
      C[(size_t)(m0 + r + 8 * hi) * NQKV + n0 + nt * 16 + lm] = (__bf16)acc[nt][r];
}

// ---- flash attention with async double-buffered K/V tiles ------------------

__global__ __launch_bounds__(256) void k_attn(const __bf16* __restrict__ QKV,
                                              const __bf16* __restrict__ Vt,
                                              __bf16* __restrict__ Z) {
  __shared__ alignas(32) __bf16 ldsK[2 * 32 * 64];        // [key][e]  2 x 4 KB
  __shared__ alignas(32) __bf16 ldsV[2 * 64 * 32];        // [e][tok]  2 x 4 KB
  __shared__ alignas(32) __bf16 sprob[8 * 16 * 32];       // per-wave probs, 8 KB
  const int tid = threadIdx.x;
  const int lane = tid & 31, wid = tid >> 5;
  const int hi = lane >> 4, lm = lane & 15, kgrp = hi * 8;
  const int qb = blockIdx.x * 128;
  const int h  = blockIdx.y;
  const int b  = blockIdx.z;
  const int g  = h >> 1;
  const int warp_q = qb + wid * 16;

  // staging assignments (uniform, 1 K-chunk + 1 V-chunk per thread)
  const int skey = tid >> 3, skc = (tid & 7) * 8;         // K: 32 rows x 8 chunks
  const int sve  = tid >> 2, svc = (tid & 3) * 8;         // V: 64 rows x 4 chunks
  const __bf16* ksrc = QKV + ((size_t)b * SEQ + skey) * NQKV + KOFF + g * D_HEAD + skc;
  const __bf16* vsrc = Vt + ((size_t)((b * N_GROUPS + g) * D_HEAD + sve)) * SEQ + svc;

  // resident Q A-fragments (16 rows x 64 e, two K=32 fragments)
  const __bf16* qrow = QKV + ((size_t)(b * SEQ + warp_q + lm)) * NQKV + h * D_HEAD;
  v16bf aq0 = load_a_frag(qrow, kgrp);
  v16bf aq1 = load_a_frag(qrow + 32, kgrp);

  float m[8], l[8];
  #pragma unroll
  for (int r = 0; r < 8; ++r) { m[r] = -INFINITY; l[r] = 0.f; }
  v8f acc[4] = {};
  __bf16* myp = sprob + wid * 512;

  const int nkb = qb / 32 + 4;                            // keys [0, qb+128)
  // prologue: stage block 0 into buffer 0
  ASYNC_CP16(ksrc, ldsK + skey * 64 + skc);
  ASYNC_CP16(vsrc, ldsV + sve * 32 + svc);

  for (int kb = 0; kb < nkb; ++kb) {
    const int kbase = kb * 32;
    const int buf = kb & 1;
    WAIT_ASYNC(0);
    __syncthreads();                                      // KV[buf] ready; KV[buf^1]+probs free
    if (kb + 1 < nkb) {
      ASYNC_CP16(ksrc + (size_t)(kbase + 32) * NQKV,
                 ldsK + (buf ^ 1) * 2048 + skey * 64 + skc);
      ASYNC_CP16(vsrc + kbase + 32,
                 ldsV + (buf ^ 1) * 2048 + sve * 32 + svc);
    }
    // scores S = Q K^T / 8 from LDS K tile
    const __bf16* bK = ldsK + buf * 2048;
    v8f s[2];
    #pragma unroll
    for (int nt = 0; nt < 2; ++nt) {
      const __bf16* krow = bK + (nt * 16 + lm) * 64 + hi * 16;
      v16bf b0 = *(const v16bf*)krow;
      v16bf b1 = *(const v16bf*)(krow + 32);
      v8f t = {};
      t = wmma_bf16(aq0, b0, t);
      t = wmma_bf16(aq1, b1, t);
      s[nt] = t;
    }
    // scale + causal mask (VALU select; EXEC stays all-ones)
    #pragma unroll
    for (int nt = 0; nt < 2; ++nt) {
      const int ki = kbase + nt * 16 + lm;
      #pragma unroll
      for (int r = 0; r < 8; ++r) {
        const int qi = warp_q + r + 8 * hi;
        float v = s[nt][r] * 0.125f;
        s[nt][r] = (ki > qi) ? -1000000.0f : v;
      }
    }
    // online softmax per C-fragment row (M = r + 8*hi)
    #pragma unroll
    for (int r = 0; r < 8; ++r) {
      float mt   = rmax16(fmaxf(s[0][r], s[1][r]));
      float mnew = fmaxf(m[r], mt);
      float fac  = __expf(m[r] - mnew);
      float p0   = __expf(s[0][r] - mnew);
      float p1   = __expf(s[1][r] - mnew);
      s[0][r] = p0; s[1][r] = p1;
      l[r] = l[r] * fac + rsum16(p0 + p1);
      m[r] = mnew;
      #pragma unroll
      for (int et = 0; et < 4; ++et) acc[et][r] *= fac;
    }
    // probs -> LDS (C layout) -> A-fragment layout
    #pragma unroll
    for (int nt = 0; nt < 2; ++nt)
      #pragma unroll
      for (int r = 0; r < 8; ++r)
        myp[(r + 8 * hi) * 32 + nt * 16 + lm] = (__bf16)s[nt][r];
    __syncthreads();
    v16bf ap = load_a_frag(myp + lm * 32, kgrp);
    // O += P V from LDS V tile (4 e-tiles of 16, K = 32 keys)
    const __bf16* bV = ldsV + buf * 2048;
    #pragma unroll
    for (int et = 0; et < 4; ++et) {
      v16bf bv = *(const v16bf*)(bV + (et * 16 + lm) * 32 + hi * 16);
      acc[et] = wmma_bf16(ap, bv, acc[et]);
    }
  }
  #pragma unroll
  for (int r = 0; r < 8; ++r) l[r] = 1.0f / l[r];
  #pragma unroll
  for (int et = 0; et < 4; ++et)
    #pragma unroll
    for (int r = 0; r < 8; ++r)
      Z[((size_t)(b * SEQ + warp_q + r + 8 * hi)) * D_MODEL + h * D_HEAD + et * 16 + lm]
          = (__bf16)(acc[et][r] * l[r]);
}

// ---- output projection with async double-buffered B tiles ------------------

__global__ __launch_bounds__(256) void k_gemm_out(const __bf16* __restrict__ Z,
                                                  const __bf16* __restrict__ BT,
                                                  const float* __restrict__ bias,
                                                  float* __restrict__ out) {
  __shared__ alignas(32) __bf16 ldsB[2 * 64 * 32];
  const int tid = threadIdx.x;
  const int lane = tid & 31, wid = tid >> 5;
  const int hi = lane >> 4, lm = lane & 15, kgrp = hi * 8;
  const int m0 = blockIdx.y * 128 + wid * 16;
  const int n0 = blockIdx.x * 64;
  const int sn = tid >> 2, sc = (tid & 3) * 8;
  const __bf16* bsrc = BT + (size_t)(n0 + sn) * D_MODEL + sc;

  v8f acc[4] = {};
  const __bf16* arow = Z + (size_t)(m0 + lm) * D_MODEL;

  ASYNC_CP16(bsrc, ldsB + sn * 32 + sc);
  for (int i = 0; i < D_MODEL / 32; ++i) {
    WAIT_ASYNC(0);
    __syncthreads();
    if (i + 1 < D_MODEL / 32)
      ASYNC_CP16(bsrc + (i + 1) * 32,
                 ldsB + ((i + 1) & 1) * 2048 + sn * 32 + sc);
    const __bf16* bufB = ldsB + (i & 1) * 2048;
    v16bf af = load_a_frag(arow + i * 32, kgrp);
    #pragma unroll
    for (int nt = 0; nt < 4; ++nt) {
      v16bf bf = *(const v16bf*)(bufB + (nt * 16 + lm) * 32 + hi * 16);
      acc[nt] = wmma_bf16(af, bf, acc[nt]);
    }
  }
  #pragma unroll
  for (int nt = 0; nt < 4; ++nt) {
    int col = n0 + nt * 16 + lm;
    float bb = bias[col];
    #pragma unroll
    for (int r = 0; r < 8; ++r)
      out[(size_t)(m0 + r + 8 * hi) * D_MODEL + col] = acc[nt][r] + bb;
  }
}

// ---- launch ----------------------------------------------------------------

extern "C" void kernel_launch(void* const* d_in, const int* in_sizes, int n_in,
                              void* d_out, int out_size, void* d_ws, size_t ws_size,
                              hipStream_t stream) {
  const float* resid = (const float*)d_in[0];
  const float* W_Q   = (const float*)d_in[1];
  const float* W_K   = (const float*)d_in[2];
  const float* W_V   = (const float*)d_in[3];
  const float* W_out = (const float*)d_in[4];
  const float* b_out = (const float*)d_in[5];
  float* out = (float*)d_out;

  char* ws = (char*)d_ws;
  __bf16* residbf = (__bf16*)(ws);                        //  8 MB [4096 x 1024]
  __bf16* WcatT   = (__bf16*)(ws + (size_t)8  * 1048576); //  4 MB [2048 x 1024]
  __bf16* WoT     = (__bf16*)(ws + (size_t)12 * 1048576); //  2 MB [1024 x 1024]
  __bf16* QKV     = (__bf16*)(ws + (size_t)14 * 1048576); // 16 MB [4096 x 2048]
  __bf16* Vt      = (__bf16*)(ws + (size_t)30 * 1048576); //  4 MB [2][8][64][2048]
  __bf16* Z       = (__bf16*)(ws + (size_t)34 * 1048576); //  8 MB [4096 x 1024]

  k_cvt_resid <<<(BS * D_MODEL) / 256, 256, 0, stream>>>(resid, residbf, BS * D_MODEL);
  k_build_wcatT<<<(2048 * 1024) / 256, 256, 0, stream>>>(W_Q, W_K, W_V, WcatT);
  k_build_woT <<<(1024 * 1024) / 256, 256, 0, stream>>>(W_out, WoT);

  k_gemm_qkv<<<dim3(NQKV / 64, BS / 128), 256, 0, stream>>>(residbf, WcatT, QKV);
  k_transpose_v<<<(BATCH * N_GROUPS * D_HEAD * SEQ) / 256, 256, 0, stream>>>(QKV, Vt);
  k_attn<<<dim3(SEQ / 128, N_HEADS, BATCH), 256, 0, stream>>>(QKV, Vt, Z);
  k_gemm_out<<<dim3(D_MODEL / 64, BS / 128), 256, 0, stream>>>(Z, WoT, b_out, out);
}
